// MHPFGT_69174743270019
// MI455X (gfx1250) — compile-verified, matrix operand order checked by reading, not run on hardware
//
#include <hip/hip_runtime.h>
#include <hip/hip_bf16.h>

// ---------------- problem constants (from reference) ----------------
#define N_NODES 50000
#define E_EDGES 400000
#define F_INF   128
#define HIDC    64
#define NHEAD   4
#define HEADD   16
#define NCLS    16
#define KHOPS   3
#define CSTEPS  1e-5f

typedef __attribute__((ext_vector_type(16))) __bf16 v16bf;
typedef __attribute__((ext_vector_type(8)))  float  v8f;

union BFrag { v16bf v; unsigned short u[16]; uint4 q[2]; };

__device__ __forceinline__ unsigned short f2bf(float f) {
    union { float f; unsigned int u; } x; x.f = f;
    unsigned int r = x.u + 0x7FFFu + ((x.u >> 16) & 1u);   // RNE
    return (unsigned short)(r >> 16);
}

// ---------------- k0a: f32 -> bf16 convert (row-major, for x) ----------------
__global__ void cvt_bf16_kernel(const float* __restrict__ src,
                                unsigned short* __restrict__ dst, int n) {
    int i = blockIdx.x * blockDim.x + threadIdx.x;
    if (i < n) dst[i] = f2bf(src[i]);
}

// ---------------- k0b: f32 weight -> bf16 in WMMA B-fragment order ----------------
// layout: [(kt*NT + nt)][lane][elem] with 16 contiguous bf16 per lane
__global__ void swizzle_w_kernel(const float* __restrict__ W,
                                 unsigned short* __restrict__ dst,
                                 int K, int Ncols) {
    int i = blockIdx.x * blockDim.x + threadIdx.x;
    if (i >= K * Ncols) return;
    int NT   = Ncols >> 4;
    int e    = i & 15;
    int lane = (i >> 4) & 31;
    int tile = i >> 9;                     // kt*NT + nt
    int nt = tile % NT, kt = tile / NT;
    int half = lane >> 4, l16 = lane & 15;
    int j = e >> 1, w = e & 1;
    int k0 = (j < 4) ? (2 * j + 8 * half) : (16 + 2 * (j - 4) + 8 * half);
    int k = kt * 32 + k0 + w;
    int n = nt * 16 + l16;
    dst[i] = f2bf(W[(size_t)k * Ncols + n]);
}

// ---------------- zero kernel (graph-capture safe scratch init) ----------------
__global__ void zero4_kernel(float4* __restrict__ p, size_t n4) {
    size_t i = (size_t)blockIdx.x * blockDim.x + threadIdx.x;
    if (i < n4) p[i] = make_float4(0.f, 0.f, 0.f, 0.f);
}

// ---------------- k1: fused h=relu(x@Wi+bi); Q,K,V projections ----------------
// block = 128 threads = 4 waves, each wave owns a 16-row tile (64 rows / block)
__global__ __launch_bounds__(128) void proj_kernel(
    const unsigned short* __restrict__ xbf,
    const unsigned short* __restrict__ Wi,   // swizzled, KT=4 NT=4
    const unsigned short* __restrict__ Wq,   // swizzled, KT=2 NT=4
    const unsigned short* __restrict__ Wk,
    const unsigned short* __restrict__ Wv,
    const float* __restrict__ bi, const float* __restrict__ bq,
    const float* __restrict__ bk, const float* __restrict__ bv,
    const float* __restrict__ temp,
    float* __restrict__ Qo, float* __restrict__ Ko,
    float* __restrict__ Vo, float* __restrict__ Ho)
{
    __shared__ __align__(16) unsigned short sWi[F_INF * HIDC];   // 16 KB (swizzled)
    __shared__ __align__(16) unsigned short sWq[HIDC * HIDC];    // 8 KB
    __shared__ __align__(16) unsigned short sWk[HIDC * HIDC];    // 8 KB
    __shared__ __align__(16) unsigned short sWv[HIDC * HIDC];    // 8 KB
    __shared__ __align__(16) unsigned short sH[4][16 * HIDC];    // 8 KB (per-wave h tile)

    const int tid = threadIdx.x;
    {   // cooperative copy (flat, coalesced b128-friendly)
        const uint4* s0 = (const uint4*)Wi;  uint4* d0 = (uint4*)sWi;
        for (int i = tid; i < F_INF * HIDC / 8; i += 128) d0[i] = s0[i];
        const uint4* s1 = (const uint4*)Wq;  uint4* d1 = (uint4*)sWq;
        const uint4* s2 = (const uint4*)Wk;  uint4* d2 = (uint4*)sWk;
        const uint4* s3 = (const uint4*)Wv;  uint4* d3 = (uint4*)sWv;
        for (int i = tid; i < HIDC * HIDC / 8; i += 128) { d1[i] = s1[i]; d2[i] = s2[i]; d3[i] = s3[i]; }
    }
    __syncthreads();

    const int wave = tid >> 5, lane = tid & 31;
    const int half = lane >> 4, l16 = lane & 15;
    const int row0 = blockIdx.x * 64 + wave * 16;
    if (row0 >= N_NODES) return;                     // N divisible by 16, whole wave skips

    // B fragment: 2x ds_load_b128 from swizzled LDS weight
    auto loadB = [&](const unsigned short* Wsw, int kt, int nt, int NT, BFrag& B) {
        const unsigned short* p = Wsw + (((kt * NT + nt) * 32 + lane) << 4);
        B.q[0] = *(const uint4*)p;
        B.q[1] = *(const uint4*)(p + 8);
    };
    // A fragment: two contiguous 8-ushort runs (ISA 7.12.2 16-bit A layout)
    auto loadA = [&](const unsigned short* rowp, BFrag& A) {
        A.q[0] = *(const uint4*)(rowp + 8 * half);
        A.q[1] = *(const uint4*)(rowp + 16 + 8 * half);
    };

    // ---- h = relu(x @ Wi + bi) : 16x128 @ 128x64
    v8f acc[4];
    #pragma unroll
    for (int nt = 0; nt < 4; ++nt) acc[nt] = {};
    #pragma unroll
    for (int ks = 0; ks < 4; ++ks) {
        BFrag A; loadA(xbf + (size_t)(row0 + l16) * F_INF + ks * 32, A);
        #pragma unroll
        for (int nt = 0; nt < 4; ++nt) {
            BFrag B; loadB(sWi, ks, nt, 4, B);
            acc[nt] = __builtin_amdgcn_wmma_f32_16x16x32_bf16(
                false, A.v, false, B.v, (short)0, acc[nt], false, false);
        }
    }
    unsigned short* h = sH[wave];
    #pragma unroll
    for (int nt = 0; nt < 4; ++nt) {
        #pragma unroll
        for (int r = 0; r < 8; ++r) {                 // C layout: row = r + 8*half, col = nt*16+l16
            int cc = nt * 16 + l16;
            float v = acc[nt][r] + bi[cc];
            h[(r + 8 * half) * HIDC + cc] = f2bf(v > 0.f ? v : 0.f);
        }
    }
    // same-wave LDS RAW: DScnt waits inserted by compiler

    // A fragments of h (16x64 -> two K-steps), 2x ds_load_b128 each
    BFrag A0, A1;
    loadA(h + l16 * HIDC, A0);
    loadA(h + l16 * HIDC + 32, A1);

    const float t0h[4] = { temp[0], temp[4], temp[8], temp[12] };  // temp[:,0], (H,K+1) row-major

    #pragma unroll
    for (int nt = 0; nt < 4; ++nt) {
        BFrag B0, B1;
        v8f cq = {}, ck = {}, cv = {};
        loadB(sWq, 0, nt, 4, B0); loadB(sWq, 1, nt, 4, B1);
        cq = __builtin_amdgcn_wmma_f32_16x16x32_bf16(false, A0.v, false, B0.v, (short)0, cq, false, false);
        cq = __builtin_amdgcn_wmma_f32_16x16x32_bf16(false, A1.v, false, B1.v, (short)0, cq, false, false);
        loadB(sWk, 0, nt, 4, B0); loadB(sWk, 1, nt, 4, B1);
        ck = __builtin_amdgcn_wmma_f32_16x16x32_bf16(false, A0.v, false, B0.v, (short)0, ck, false, false);
        ck = __builtin_amdgcn_wmma_f32_16x16x32_bf16(false, A1.v, false, B1.v, (short)0, ck, false, false);
        loadB(sWv, 0, nt, 4, B0); loadB(sWv, 1, nt, 4, B1);
        cv = __builtin_amdgcn_wmma_f32_16x16x32_bf16(false, A0.v, false, B0.v, (short)0, cv, false, false);
        cv = __builtin_amdgcn_wmma_f32_16x16x32_bf16(false, A1.v, false, B1.v, (short)0, cv, false, false);

        #pragma unroll
        for (int r = 0; r < 8; ++r) {
            int cc = nt * 16 + l16;
            size_t node = (size_t)(row0 + r + 8 * half);
            float q = cq[r] + bq[cc];
            float k = ck[r] + bk[cc];
            float v = cv[r] + bv[cc];
            Qo[node * HIDC + cc] = q > 0.f ? 1.f + q : __expf(q);   // 1 + elu
            Ko[node * HIDC + cc] = k > 0.f ? 1.f + k : __expf(k);
            Vo[node * HIDC + cc] = v;
            Ho[node * HIDC + cc] = v * t0h[nt];                     // head == nt
        }
    }
}

// ---------------- k2: M0[n,h,d,c] = Kt[n,h,d] * V[n,h,c] (float4) ----------------
__global__ void initM_kernel(const float* __restrict__ Kt, const float* __restrict__ V,
                             float* __restrict__ M) {
    size_t i = (size_t)blockIdx.x * blockDim.x + threadIdx.x;   // one float4 each
    if (i >= (size_t)N_NODES * 256) return;
    size_t n = i >> 8; int rem = (int)(i & 255);
    int hh = rem >> 6, d = (rem >> 2) & 15, c4 = (rem & 3) * 4;
    float  kk = Kt[n * HIDC + hh * 16 + d];
    float4 v  = *(const float4*)(V + n * HIDC + hh * 16 + c4);
    float4 o  = make_float4(kk * v.x, kk * v.y, kk * v.z, kk * v.w);
    *(float4*)(M + n * 1024 + hh * 256 + d * 16 + c4) = o;
}

// ---------------- k3: per-edge scatter-add (segment_sum) ----------------
// one 256-thread block per edge; contiguous 4KB b128 row read, coalesced f32 atomics
__global__ __launch_bounds__(256) void edge_kernel(
    const int* __restrict__ ei,
    const float* __restrict__ Mold, const float* __restrict__ Ktold,
    float* __restrict__ Mnew, float* __restrict__ Ktnew)
{
    int e = blockIdx.x, tid = threadIdx.x;
    int s = ei[e], d = ei[E_EDGES + e];
    const float* Ms = Mold + (size_t)s * 1024;
    float*       Md = Mnew + (size_t)d * 1024;
    __builtin_prefetch(Ktold + (size_t)s * HIDC, 0, 1);   // global_prefetch
    float4 v = *(const float4*)(Ms + tid * 4);
    atomicAdd(&Md[tid * 4 + 0], v.x);
    atomicAdd(&Md[tid * 4 + 1], v.y);
    atomicAdd(&Md[tid * 4 + 2], v.z);
    atomicAdd(&Md[tid * 4 + 3], v.w);
    if (tid < 16) {
        float4 kv = *(const float4*)(Ktold + (size_t)s * HIDC + tid * 4);
        float* kd = Ktnew + (size_t)d * HIDC + tid * 4;
        atomicAdd(kd + 0, kv.x);
        atomicAdd(kd + 1, kv.y);
        atomicAdd(kd + 2, kv.z);
        atomicAdd(kd + 3, kv.w);
    }
}

// ---------------- k4: Hm = Q.M, den = Q.Kt, hidden += gamma*Hm/den ----------------
// one thread per (node, head, c-quad): all M traffic as b128
__global__ void combine_kernel(const float* __restrict__ Q, const float* __restrict__ M,
                               const float* __restrict__ Kt,
                               const float* __restrict__ hopwise,
                               const float* __restrict__ temp,
                               float* __restrict__ Hid, int hop)
{
    size_t i = (size_t)blockIdx.x * blockDim.x + threadIdx.x;
    if (i >= (size_t)N_NODES * NHEAD * 4) return;
    int c4 = (int)(i & 3) * 4;
    int hh = (int)((i >> 2) & 3);
    size_t n = i >> 4;
    const float* q = Q  + n * HIDC + hh * 16;
    const float* k = Kt + n * HIDC + hh * 16;
    const float* m = M  + n * 1024 + hh * 256 + c4;
    float4 hm = make_float4(0.f, 0.f, 0.f, 0.f);
    float den = 0.f;
    #pragma unroll
    for (int d = 0; d < 16; ++d) {
        float qq = q[d];
        float4 mv = *(const float4*)(m + d * 16);
        hm.x += qq * mv.x; hm.y += qq * mv.y; hm.z += qq * mv.z; hm.w += qq * mv.w;
        den += qq * k[d];
    }
    den += CSTEPS;
    // layerwise = softmax(temp, axis=0) (over heads), column hop+1
    int col = hop + 1;
    float e0 = __expf(temp[0 * 4 + col]), e1 = __expf(temp[1 * 4 + col]);
    float e2 = __expf(temp[2 * 4 + col]), e3 = __expf(temp[3 * 4 + col]);
    float lw = (hh == 0 ? e0 : hh == 1 ? e1 : hh == 2 ? e2 : e3) / (e0 + e1 + e2 + e3);
    float g = hopwise[col] * lw / den;
    float4* hp = (float4*)(Hid + n * HIDC + hh * 16 + c4);
    float4 hv = *hp;
    hv.x += g * hm.x; hv.y += g * hm.y; hv.z += g * hm.z; hv.w += g * hm.w;
    *hp = hv;
}

// ---------------- k5: out = hidden @ Wo + bo (16x64 @ 64x16 per wave tile) ----------------
__global__ __launch_bounds__(128) void out_kernel(
    const float* __restrict__ Hid, const unsigned short* __restrict__ Wo,  // swizzled KT=2 NT=1
    const float* __restrict__ bo, float* __restrict__ out)
{
    __shared__ __align__(16) unsigned short sWo[HIDC * NCLS];   // 2 KB (swizzled)
    __shared__ __align__(16) unsigned short sH[4][16 * HIDC];   // 8 KB
    const int tid = threadIdx.x;
    {
        const uint4* s0 = (const uint4*)Wo;  uint4* d0 = (uint4*)sWo;
        for (int i = tid; i < HIDC * NCLS / 8; i += 128) d0[i] = s0[i];
    }
    __syncthreads();

    const int wave = tid >> 5, lane = tid & 31;
    const int half = lane >> 4, l16 = lane & 15;
    const int row0 = blockIdx.x * 64 + wave * 16;
    if (row0 >= N_NODES) return;

    unsigned short* h = sH[wave];
    for (int i = lane; i < 16 * HIDC; i += 32) {
        int r = i >> 6, c = i & 63;
        h[i] = f2bf(Hid[(size_t)(row0 + r) * HIDC + c]);
    }

    BFrag A0, A1, B0, B1;
    A0.q[0] = *(const uint4*)(h + l16 * HIDC + 8 * half);
    A0.q[1] = *(const uint4*)(h + l16 * HIDC + 16 + 8 * half);
    A1.q[0] = *(const uint4*)(h + l16 * HIDC + 32 + 8 * half);
    A1.q[1] = *(const uint4*)(h + l16 * HIDC + 48 + 8 * half);
    {
        const unsigned short* p0 = sWo + (lane << 4);            // kt=0
        const unsigned short* p1 = sWo + ((32 + lane) << 4);     // kt=1
        B0.q[0] = *(const uint4*)p0; B0.q[1] = *(const uint4*)(p0 + 8);
        B1.q[0] = *(const uint4*)p1; B1.q[1] = *(const uint4*)(p1 + 8);
    }
    v8f c = {};
    c = __builtin_amdgcn_wmma_f32_16x16x32_bf16(false, A0.v, false, B0.v, (short)0, c, false, false);
    c = __builtin_amdgcn_wmma_f32_16x16x32_bf16(false, A1.v, false, B1.v, (short)0, c, false, false);
    #pragma unroll
    for (int r = 0; r < 8; ++r)
        out[(size_t)(row0 + r + 8 * half) * NCLS + l16] = c[r] + bo[l16];
}

// ---------------- host side ----------------
static inline size_t alignUp(size_t v) { return (v + 255) & ~(size_t)255; }

extern "C" void kernel_launch(void* const* d_in, const int* in_sizes, int n_in,
                              void* d_out, int out_size, void* d_ws, size_t ws_size,
                              hipStream_t stream) {
    (void)in_sizes; (void)n_in; (void)out_size; (void)ws_size;
    const float* x   = (const float*)d_in[0];
    const int*   ei  = (const int*)  d_in[1];
    const float* Wi  = (const float*)d_in[2];  const float* bi = (const float*)d_in[3];
    const float* Wq  = (const float*)d_in[4];  const float* bq = (const float*)d_in[5];
    const float* Wk  = (const float*)d_in[6];  const float* bk = (const float*)d_in[7];
    const float* Wv  = (const float*)d_in[8];  const float* bv = (const float*)d_in[9];
    const float* Wo  = (const float*)d_in[10]; const float* bo = (const float*)d_in[11];
    const float* hw  = (const float*)d_in[12]; const float* temp = (const float*)d_in[13];
    float* out = (float*)d_out;

    // workspace carve-up
    char* ws = (char*)d_ws; size_t off = 0;
    auto carve = [&](size_t bytes) { char* p = ws + off; off = alignUp(off + bytes); return p; };
    unsigned short* xbf  = (unsigned short*)carve((size_t)N_NODES * F_INF * 2);
    unsigned short* Wib  = (unsigned short*)carve((size_t)F_INF * HIDC * 2);
    unsigned short* Wqb  = (unsigned short*)carve((size_t)HIDC * HIDC * 2);
    unsigned short* Wkb  = (unsigned short*)carve((size_t)HIDC * HIDC * 2);
    unsigned short* Wvb  = (unsigned short*)carve((size_t)HIDC * HIDC * 2);
    unsigned short* Wob  = (unsigned short*)carve((size_t)HIDC * NCLS * 2);
    float* Qb  = (float*)carve((size_t)N_NODES * HIDC * 4);
    float* Vb  = (float*)carve((size_t)N_NODES * HIDC * 4);
    float* Hid = (float*)carve((size_t)N_NODES * HIDC * 4);
    float* KtA = (float*)carve((size_t)N_NODES * HIDC * 4);
    float* KtB = (float*)carve((size_t)N_NODES * HIDC * 4);
    float* MA  = (float*)carve((size_t)N_NODES * 1024 * 4);
    float* MB  = (float*)carve((size_t)N_NODES * 1024 * 4);

    // k0: conversions (x row-major bf16; weights fragment-swizzled bf16)
    cvt_bf16_kernel<<<(N_NODES * F_INF + 255) / 256, 256, 0, stream>>>(x, xbf, N_NODES * F_INF);
    swizzle_w_kernel<<<(F_INF * HIDC + 255) / 256, 256, 0, stream>>>(Wi, Wib, F_INF, HIDC);
    swizzle_w_kernel<<<(HIDC * HIDC + 255) / 256, 256, 0, stream>>>(Wq, Wqb, HIDC, HIDC);
    swizzle_w_kernel<<<(HIDC * HIDC + 255) / 256, 256, 0, stream>>>(Wk, Wkb, HIDC, HIDC);
    swizzle_w_kernel<<<(HIDC * HIDC + 255) / 256, 256, 0, stream>>>(Wv, Wvb, HIDC, HIDC);
    swizzle_w_kernel<<<(HIDC * NCLS + 255) / 256, 256, 0, stream>>>(Wo, Wob, HIDC, NCLS);

    // k1: fused projections (+ hidden hop-0 init)
    proj_kernel<<<(N_NODES + 63) / 64, 128, 0, stream>>>(
        xbf, Wib, Wqb, Wkb, Wvb, bi, bq, bk, bv, temp, Qb, KtA, Vb, Hid);

    // k2: M0 outer products
    initM_kernel<<<(N_NODES * 256 + 255) / 256, 256, 0, stream>>>(KtA, Vb, MA);

    // hops
    float *Mc = MA, *Mn = MB, *Kc = KtA, *Kn = KtB;
    const size_t Mn4 = (size_t)N_NODES * 1024 / 4;
    const size_t Kn4 = (size_t)N_NODES * HIDC / 4;
    for (int hop = 0; hop < KHOPS; ++hop) {
        zero4_kernel<<<(unsigned)((Mn4 + 255) / 256), 256, 0, stream>>>((float4*)Mn, Mn4);
        zero4_kernel<<<(unsigned)((Kn4 + 255) / 256), 256, 0, stream>>>((float4*)Kn, Kn4);
        edge_kernel<<<E_EDGES, 256, 0, stream>>>(ei, Mc, Kc, Mn, Kn);
        combine_kernel<<<(N_NODES * NHEAD * 4 + 255) / 256, 256, 0, stream>>>(
            Qb, Mn, Kn, hw, temp, Hid, hop);
        float* t;
        t = Mc; Mc = Mn; Mn = t;
        t = Kc; Kc = Kn; Kn = t;
    }

    // k5: output projection
    out_kernel<<<(N_NODES + 63) / 64, 128, 0, stream>>>(Hid, Wob, bo, out);
}
